// LMLIFSNN_78443282694355
// MI455X (gfx1250) — matrix operand order repaired and checked by reading, not compiled, not run on hardware
//
#include <hip/hip_runtime.h>
#include <hip/hip_bf16.h>

typedef __attribute__((ext_vector_type(16))) _Float16 v16h;
typedef __attribute__((ext_vector_type(8)))  _Float16 v8h;
typedef __attribute__((ext_vector_type(4)))  _Float16 v4h;
typedef __attribute__((ext_vector_type(8)))  float    v8f;

#define LIF_THETA 1.0f
#define LIF_EPS   1e-5f

// ---------------------------------------------------------------------------
// f32 -> f16 conversion, 4 elements / thread (b128 load, b64 store)
// ---------------------------------------------------------------------------
__global__ void cvt_f32_f16(const float* __restrict__ s, _Float16* __restrict__ d, int n4) {
    int i = blockIdx.x * blockDim.x + threadIdx.x;
    if (i < n4) {
        float4 v = ((const float4*)s)[i];
        v4h o = { (_Float16)v.x, (_Float16)v.y, (_Float16)v.z, (_Float16)v.w };
        ((v4h*)d)[i] = o;
    }
}

__global__ void zero_f32(float* __restrict__ p, int n) {
    int i = blockIdx.x * blockDim.x + threadIdx.x;
    if (i < n) p[i] = 0.0f;
}

// ---------------------------------------------------------------------------
// CDNA5 async Global -> LDS copy (16 B per lane), tracked by ASYNCcnt.
//   LDS byte address taken from the vdst VGPR; global addr = VGPR pair (GV mode).
// ---------------------------------------------------------------------------
__device__ __forceinline__ void async_ld16(unsigned lds_off, const void* g) {
    asm volatile("global_load_async_to_lds_b128 %0, %1, off"
                 :: "v"(lds_off), "v"((unsigned long long)(uintptr_t)g)
                 : "memory");
}
__device__ __forceinline__ void wait_async0() {
    asm volatile("s_wait_asynccnt 0x0" ::: "memory");
}

// ---------------------------------------------------------------------------
// GEMM + fused BN-stats: C[M,N] (f32) = A[M,K] f16 * W[N,K]^T f16,
// plus per-column atomic sum / sum-of-squares from the accumulators.
//
// Wave32 WMMA f32_16x16x32_f16. Block = 256 threads = 8 waves (2M x 4N),
// wave tile 64x64 (4x4 accumulators), block tile 128(M) x 256(N).
//
// Data path: double-buffered async Global->LDS staging of the 128x32 A panel
// (8 KB) and 256x32 B panel (16 KB) per k-step; WMMA operands read from LDS.
// Rows padded to 80 B so every ds_load_b128 is 2-phase conflict-free
// (5*l mod 16 is a permutation -> 16 distinct bank-quads per half-wave).
//
// Grid = (N/256 fast, M/128 slow): consecutive blocks share one A panel
// (L2-resident) and W (8 MB f16) stays resident in the 192 MB L2, so A
// streams from HBM exactly once.
// ---------------------------------------------------------------------------
#define A_PANEL   10240u            // 128 rows * 80 B
#define B_PANEL   20480u            // 256 rows * 80 B
#define SMEM_BYTES (2u * (A_PANEL + B_PANEL))   // 61440 B

__launch_bounds__(256)
__global__ void gemm_f16_wmma(const _Float16* __restrict__ A,
                              const _Float16* __restrict__ W,
                              float* __restrict__ C,
                              float* __restrict__ sum,
                              float* __restrict__ sumsq,
                              int M, int N, int K) {
    __shared__ __align__(128) char smem[SMEM_BYTES];
    // LDS aperture keeps the byte offset in addr[31:0] (ISA 10.2)
    const unsigned lds_base = (unsigned)(uintptr_t)(&smem[0]);

    const int t     = threadIdx.x;
    const int wave  = t >> 5;
    const int lane  = t & 31;
    const int lrow  = lane & 15;
    const int lhalf = lane >> 4;
    const int wm    = wave & 1;    // 2 M-strips of 64
    const int wn    = wave >> 1;   // 4 N-strips of 64

    const int base_m = blockIdx.y * 128;
    const int base_n = blockIdx.x * 256;
    const int m0 = base_m + wm * 64;
    const int n0 = base_n + wn * 64;

    v8f acc[4][4];
    #pragma unroll
    for (int mi = 0; mi < 4; ++mi)
        #pragma unroll
        for (int ni = 0; ni < 4; ++ni)
            acc[mi][ni] = (v8f){};

    // ---- async-copy source/destination for this thread ----
    // Each k-step panel row = 64 B = 4 chunks of 16 B; thread t owns
    // chunk part (t&3) of row (t>>2) and the rows +64/(+128,+192 for B).
    const int crow = t >> 2, cpart = t & 3;
    const _Float16* gA = A + (size_t)(base_m + crow) * K + cpart * 8;
    const _Float16* gB = W + (size_t)(base_n + crow) * K + cpart * 8;
    const unsigned lchunk = (unsigned)(crow * 80 + cpart * 16);

    // ---- LDS read offsets for WMMA fragments ----
    // A frag (ISA 7.12.2 16-bit A 16x32): 16 B chunks at +0 / +32 from
    //   row*80 + lhalf*16.   B frag: same, 16 B chunks at +0 / +32 from row*80 + lhalf*16.
    const unsigned aOff = (unsigned)((wm * 64 + lrow) * 80 + lhalf * 16);
    const unsigned bOff = (unsigned)((wn * 64 + lrow) * 80 + lhalf * 16);

    const int nk = K >> 5;

    // prologue: fill buffer 0
    {
        unsigned a0 = lds_base + lchunk;
        unsigned b0 = lds_base + 2u * A_PANEL + lchunk;
        async_ld16(a0,           gA);
        async_ld16(a0 + 5120u,   gA + (size_t)64 * K);
        async_ld16(b0,           gB);
        async_ld16(b0 + 5120u,   gB + (size_t)64 * K);
        async_ld16(b0 + 10240u,  gB + (size_t)128 * K);
        async_ld16(b0 + 15360u,  gB + (size_t)192 * K);
        gA += 32; gB += 32;
    }
    wait_async0();
    __syncthreads();

    for (int ks = 0; ks < nk; ++ks) {
        const int cur = ks & 1;

        // issue async fill of the other buffer (overlapped with compute)
        if (ks + 1 < nk) {
            unsigned a0 = lds_base + (cur ^ 1) * A_PANEL + lchunk;
            unsigned b0 = lds_base + 2u * A_PANEL + (cur ^ 1) * B_PANEL + lchunk;
            async_ld16(a0,           gA);
            async_ld16(a0 + 5120u,   gA + (size_t)64 * K);
            async_ld16(b0,           gB);
            async_ld16(b0 + 5120u,   gB + (size_t)64 * K);
            async_ld16(b0 + 10240u,  gB + (size_t)128 * K);
            async_ld16(b0 + 15360u,  gB + (size_t)192 * K);
            gA += 32; gB += 32;
        }

        // compute from buffer `cur`
        const char* sA = smem + cur * A_PANEL + aOff;
        const char* sB = smem + 2u * A_PANEL + cur * B_PANEL + bOff;

        v16h af[4], bf[4];
        #pragma unroll
        for (int i = 0; i < 4; ++i) {
            v8h alo = *(const v8h*)(sA + i * 1280);
            v8h ahi = *(const v8h*)(sA + i * 1280 + 32);
            af[i] = __builtin_shufflevector(alo, ahi, 0,1,2,3,4,5,6,7,8,9,10,11,12,13,14,15);
            v8h blo = *(const v8h*)(sB + i * 1280);
            v8h bhi = *(const v8h*)(sB + i * 1280 + 32);
            bf[i] = __builtin_shufflevector(blo, bhi, 0,1,2,3,4,5,6,7,8,9,10,11,12,13,14,15);
        }
        #pragma unroll
        for (int mi = 0; mi < 4; ++mi)
            #pragma unroll
            for (int ni = 0; ni < 4; ++ni)
                acc[mi][ni] = __builtin_amdgcn_wmma_f32_16x16x32_f16(
                    false, af[mi], false, bf[ni], (short)0, acc[mi][ni], false, false);

        // protects both: next iter reads buf cur^1 (writes done) and next
        // iter's async overwrite of buf cur (all waves finished reading it)
        wait_async0();
        __syncthreads();
    }

    // D layout (ISA 7.12.2): VGPR r, lanes 0-15 -> M=r, lanes 16-31 -> M=r+8
    #pragma unroll
    for (int mi = 0; mi < 4; ++mi) {
        float* c0 = C + (size_t)(m0 + 16 * mi + lhalf * 8) * N + n0 + lrow;
        #pragma unroll
        for (int ni = 0; ni < 4; ++ni)
            #pragma unroll
            for (int r = 0; r < 8; ++r)
                c0[(size_t)r * N + ni * 16] = acc[mi][ni][r];
    }

    // fused BN statistics: lane owns column (n0 + ni*16 + lrow) for 32 rows
    #pragma unroll
    for (int ni = 0; ni < 4; ++ni) {
        float s = 0.0f, q = 0.0f;
        #pragma unroll
        for (int mi = 0; mi < 4; ++mi)
            #pragma unroll
            for (int r = 0; r < 8; ++r) {
                float v = acc[mi][ni][r];
                s += v;
                q += v * v;
            }
        int col = n0 + 16 * ni + lrow;
        unsafeAtomicAdd(&sum[col], s);
        unsafeAtomicAdd(&sumsq[col], q);
    }
}

// ---------------------------------------------------------------------------
// Fused BatchNorm-apply + sigmoid(beta) + LIF scan over T.
// One thread per (b, channel); coalesced per-timestep row reads.
// ---------------------------------------------------------------------------
__global__ void bn_lif(const float* __restrict__ h, const float* __restrict__ sum,
                       const float* __restrict__ sumsq,
                       const float* __restrict__ gamma, const float* __restrict__ bias,
                       const float* __restrict__ beta_p, const float* __restrict__ U0,
                       _Float16* __restrict__ spk_h, float* __restrict__ spk_f,
                       int B, int T, int N) {
    int idx = blockIdx.x * blockDim.x + threadIdx.x;
    int i = idx % N;
    int b = idx / N;
    if (b >= B) return;

    float invM = 1.0f / (float)(B * T);
    float mu   = sum[i] * invM;
    float var  = sumsq[i] * invM - mu * mu;        // biased variance (training BN)
    float sc   = gamma[i] * rsqrtf(var + LIF_EPS);
    float sh   = bias[i] - mu * sc;
    float beta = 1.0f / (1.0f + expf(-beta_p[i]));
    float omb  = 1.0f - beta;

    float U = U0[(size_t)b * N + i];
    float S = 0.0f;

    const float* hp = h     + (size_t)b * T * N + i;
    _Float16*    op = spk_h + (size_t)b * T * N + i;
    float*       of = spk_f ? spk_f + (size_t)b * T * N + i : nullptr;

    for (int t = 0; t < T; ++t) {
        float xv = hp[(size_t)t * N] * sc + sh;
        U = beta * (U - LIF_THETA * S) + omb * xv;  // soft reset
        S = (U - LIF_THETA >= 0.0f) ? 1.0f : 0.0f;  // heaviside spike
        op[(size_t)t * N] = (_Float16)S;
        if (of) of[(size_t)t * N] = S;
    }
}

// ---------------------------------------------------------------------------
extern "C" void kernel_launch(void* const* d_in, const int* in_sizes, int n_in,
                              void* d_out, int out_size, void* d_ws, size_t ws_size,
                              hipStream_t stream) {
    (void)in_sizes; (void)n_in; (void)out_size; (void)ws_size;

    const int B = 32, T = 1024, J = 1024, H = 2048;
    const int M = B * T;  // 32768

    const float* x = (const float*)d_in[0];
    const float* Wl[3]    = { (const float*)d_in[1], (const float*)d_in[6],  (const float*)d_in[11] };
    const float* betal[3] = { (const float*)d_in[2], (const float*)d_in[7],  (const float*)d_in[12] };
    const float* gammal[3]= { (const float*)d_in[3], (const float*)d_in[8],  (const float*)d_in[13] };
    const float* biasl[3] = { (const float*)d_in[4], (const float*)d_in[9],  (const float*)d_in[14] };
    const float* U0l[3]   = { (const float*)d_in[5], (const float*)d_in[10], (const float*)d_in[15] };
    const int Kl[3] = { J, H, H };
    const int Nl[3] = { H, H, J };

    // workspace layout
    char* ws = (char*)d_ws;
    float*    hbuf = (float*)ws;                                  // 32768*2048 f32 = 256 MB
    _Float16* abuf = (_Float16*)(ws + (size_t)268435456);         // 32768*2048 f16 = 128 MB
    _Float16* wbuf = (_Float16*)(ws + (size_t)402653184);         // 2048*2048  f16 =   8 MB
    float*    sum  = (float*)(ws + (size_t)411041792);            // 2048 f32
    float*    sumsq= sum + 2048;                                  // 2048 f32

    // x (f32) -> A matrix (f16)
    {
        int n4 = M * J / 4;
        cvt_f32_f16<<<dim3((n4 + 255) / 256), dim3(256), 0, stream>>>(x, abuf, n4);
    }

    for (int l = 0; l < 3; ++l) {
        const int K = Kl[l], N = Nl[l];
        const bool last = (l == 2);

        // W (f32 [N,K]) -> f16
        int nw4 = N * K / 4;
        cvt_f32_f16<<<dim3((nw4 + 255) / 256), dim3(256), 0, stream>>>(Wl[l], wbuf, nw4);

        // zero stats, then GEMM with fused per-column sum/sumsq atomics
        zero_f32<<<dim3((4096 + 255) / 256), dim3(256), 0, stream>>>(sum, 4096);
        gemm_f16_wmma<<<dim3(N / 256, M / 128), dim3(256), 0, stream>>>(
            abuf, wbuf, hbuf, sum, sumsq, M, N, K);

        // BN apply + LIF scan; spikes -> abuf (f16) for next layer, f32 out on last
        bn_lif<<<dim3((B * N) / 256), dim3(256), 0, stream>>>(
            hbuf, sum, sumsq, gammal[l], biasl[l], betal[l], U0l[l],
            abuf, last ? (float*)d_out : nullptr, B, T, N);
    }
}
